// Attention_33354716021494
// MI455X (gfx1250) — matrix operand order repaired and checked by reading, not compiled, not run on hardware
//
#include <hip/hip_runtime.h>
#include <hip/hip_bf16.h>

// MI455X / gfx1250 fused causal multi-head attention layer.
// - All matmuls on v_wmma_f32_16x16x32_f16 (f16 operands, f32 accumulate).
// - GEMM operand tiles staged into LDS by the Tensor Data Mover
//   (tensor_load_to_lds, TENSORcnt, double-buffered).
// - Attention is flash-style (online softmax): the [B,H,S,S] score matrix
//   (~0.54 GB as f32) is never materialized.

typedef _Float16 half_t;
typedef __attribute__((ext_vector_type(16))) _Float16 v16h;
typedef __attribute__((ext_vector_type(8)))  _Float16 v8h;
typedef __attribute__((ext_vector_type(8)))  float    v8f;
typedef __attribute__((ext_vector_type(4)))  unsigned int u32x4;
typedef __attribute__((ext_vector_type(8)))  unsigned int u32x8;

#define WMMA_F16(a, b, c) \
  __builtin_amdgcn_wmma_f32_16x16x32_f16(false, (a), false, (b), (short)0, (c), false, false)

// ---- Tensor Data Mover: 2D tile load Global -> LDS (ISA ch. 8 D# layout) ----
// 2-group descriptor form (VADDR2/3 NULL -> tensors up to 2D).
__device__ __forceinline__ unsigned int lds_byte_addr(const void* p) {
  // Flat shared-aperture address: addr[31:0] is the LDS byte offset (ISA 10.2).
  return (unsigned int)(size_t)p;
}

__device__ __forceinline__ void tdm_load_tile_2d(unsigned int lds_addr,
                                                 const half_t* gptr,
                                                 unsigned int k_len,   // tensor_dim0 (elems, fastest)
                                                 unsigned int n_len,   // tensor_dim1 (rows)
                                                 unsigned int tile_k,  // tile_dim0
                                                 unsigned int tile_n)  // tile_dim1
{
  const unsigned long long ga = (unsigned long long)(size_t)gptr;
  u32x4 g0;
  g0[0] = 1u;                                                   // count=1, user desc
  g0[1] = lds_addr;                                             // lds_addr [63:32]
  g0[2] = (unsigned int)ga;                                     // global_addr [95:64]
  g0[3] = ((unsigned int)(ga >> 32) & 0x01FFFFFFu) | 0x80000000u; // [120:96] + type=2
  u32x8 g1;
  g1[0] = 0x00010000u;                                          // data_size=1 (2 bytes)
  g1[1] = (k_len & 0xFFFFu) << 16;                              // tensor_dim0 lo
  g1[2] = ((k_len >> 16) & 0xFFFFu) | ((n_len & 0xFFFFu) << 16);// dim0 hi | dim1 lo
  g1[3] = ((n_len >> 16) & 0xFFFFu) | (tile_k << 16);           // dim1 hi | tile_dim0
  g1[4] = tile_n & 0xFFFFu;                                     // tile_dim1 | tile_dim2=0
  g1[5] = k_len;                                                // tensor_dim0_stride lo32
  g1[6] = 0u;                                                   // stride hi | dim1_stride lo
  g1[7] = 0u;
  asm volatile("tensor_load_to_lds %0, %1" :: "s"(g0), "s"(g1) : "memory");
}

// ---- WMMA fragment loaders (contiguous k = 16h + s order, used for A and B alike) ------------
// A (16x32, MxK): lane m = l&15; lanes 0-15 hold K 0-15, lanes 16-31
// hold K 16-31 -> two 16B contiguous loads per lane.
__device__ __forceinline__ v16h load_a_frag(const half_t* __restrict__ A, int lda,
                                            int m0, int k0, int lane) {
  const int m  = m0 + (lane & 15);
  const int kb = k0 + ((lane >> 4) << 4);
  const half_t* p = A + (size_t)m * lda + kb;
  v8h lo = *(const v8h*)(p);
  v8h hi = *(const v8h*)(p + 8);
  v16h r;
#pragma unroll
  for (int j = 0; j < 8; ++j) { r[j] = lo[j]; r[8 + j] = hi[j]; }
  return r;
}

// B (32x16, KxN) with source stored as [N, K] row-major (K contiguous):
// lane n = l&15 reads 16 contiguous halfs, lanes 0-15 cover K 0-15,
// lanes 16-31 cover K 16-31.
__device__ __forceinline__ v16h load_b_frag(const half_t* __restrict__ Bm, int ldb,
                                            int n0, int k0, int lane) {
  const int n  = n0 + (lane & 15);
  const int kb = k0 + ((lane >> 4) << 4);
  return *(const v16h*)(Bm + (size_t)n * ldb + kb);
}

// LDS-tile variants: tile stored as [rows][32] halfs (row stride 64B).
__device__ __forceinline__ v16h lds_a_frag(const half_t* base, int m0, int lane) {
  const int m  = m0 + (lane & 15);
  const int kb = (lane >> 4) << 4;
  const half_t* p = base + m * 32 + kb;
  v8h lo = *(const v8h*)(p);
  v8h hi = *(const v8h*)(p + 8);
  v16h r;
#pragma unroll
  for (int j = 0; j < 8; ++j) { r[j] = lo[j]; r[8 + j] = hi[j]; }
  return r;
}
__device__ __forceinline__ v16h lds_b_frag(const half_t* base, int n0, int lane) {
  const int n  = n0 + (lane & 15);
  const int kb = (lane >> 4) << 4;
  return *(const v16h*)(base + n * 32 + kb);
}

// ---- f32 -> f16 convert -----------------------------------------------------
__global__ __launch_bounds__(256) void f32_to_f16_kernel(const float* __restrict__ in,
                                                         half_t* __restrict__ out,
                                                         size_t n) {
  size_t i = (size_t)blockIdx.x * blockDim.x + threadIdx.x;
  if (i < n) out[i] = (half_t)in[i];
}

// ---- GEMM: C[M,N] = A[M,K] * B[N,K]^T  (both K-major) -----------------------
// 256 threads = 8 waves; wave tile 32x32, workgroup 64x128. TDM stages the
// 64x32 A-slab and 128x32 B-slab into LDS, double-buffered over K.
template <bool STORE_F32>
__global__ __launch_bounds__(256) void gemm_wmma_kernel(const half_t* __restrict__ A,
                                                        const half_t* __restrict__ Bm,
                                                        void* __restrict__ Cout,
                                                        int M, int N, int K) {
  __shared__ half_t As[2][64 * 32];    // 8 KB
  __shared__ half_t Bs[2][128 * 32];   // 16 KB

  const int lane = threadIdx.x;               // 0..31
  const int wave = threadIdx.y;               // 0..7 (2 x 4 wave grid)
  const int m0b = blockIdx.y * 64;
  const int n0b = blockIdx.x * 128;
  const int wm = (wave >> 2) * 32;
  const int wn = (wave & 3) * 32;

  // Prime buffer 0 via the Tensor Data Mover (one wave issues; EXEC ignored).
  if (wave == 0) {
    tdm_load_tile_2d(lds_byte_addr(&As[0][0]), A  + (size_t)m0b * K, K, M, 32, 64);
    tdm_load_tile_2d(lds_byte_addr(&Bs[0][0]), Bm + (size_t)n0b * K, K, N, 32, 128);
    __builtin_amdgcn_s_wait_tensorcnt(0);
  }
  __syncthreads();

  v8f acc00 = {}, acc01 = {}, acc10 = {}, acc11 = {};
  const int nsteps = K >> 5;
  for (int step = 0; step < nsteps; ++step) {
    const int cur = step & 1;
    const bool more = (step + 1) < nsteps;
    if (wave == 0 && more) {
      const int k0n = (step + 1) << 5;
      tdm_load_tile_2d(lds_byte_addr(&As[cur ^ 1][0]), A  + (size_t)m0b * K + k0n, K, M, 32, 64);
      tdm_load_tile_2d(lds_byte_addr(&Bs[cur ^ 1][0]), Bm + (size_t)n0b * K + k0n, K, N, 32, 128);
    }

    v16h a0 = lds_a_frag(&As[cur][0], wm,      lane);
    v16h a1 = lds_a_frag(&As[cur][0], wm + 16, lane);
    v16h b0 = lds_b_frag(&Bs[cur][0], wn,      lane);
    v16h b1 = lds_b_frag(&Bs[cur][0], wn + 16, lane);
    acc00 = WMMA_F16(a0, b0, acc00);
    acc01 = WMMA_F16(a0, b1, acc01);
    acc10 = WMMA_F16(a1, b0, acc10);
    acc11 = WMMA_F16(a1, b1, acc11);

    if (wave == 0 && more) __builtin_amdgcn_s_wait_tensorcnt(0);
    __syncthreads();  // next buffer ready / this buffer free for TDM overwrite
  }

  // C/D layout: lane holds col = l&15, rows (l>>4)*8 + r.
  const int col = lane & 15;
  const int rh  = (lane >> 4) * 8;
  const int m0 = m0b + wm;
  const int n0 = n0b + wn;
  auto store = [&](int m, int n, float v) {
    if constexpr (STORE_F32) ((float*)Cout)[(size_t)m * N + n] = v;
    else                     ((half_t*)Cout)[(size_t)m * N + n] = (half_t)v;
  };
#pragma unroll
  for (int r = 0; r < 8; ++r) {
    store(m0 + rh + r,      n0 + col,      acc00[r]);
    store(m0 + rh + r,      n0 + 16 + col, acc01[r]);
    store(m0 + 16 + rh + r, n0 + col,      acc10[r]);
    store(m0 + 16 + rh + r, n0 + 16 + col, acc11[r]);
  }
}

// ---- RoPE + head split ------------------------------------------------------
// qkv [B,S,3D] f16 -> Q,K [B,H,S,hd] (rotated), V transposed [B,H,hd,S].
__global__ __launch_bounds__(256) void rope_split_kernel(const half_t* __restrict__ qkv,
                                                         half_t* __restrict__ Qo,
                                                         half_t* __restrict__ Ko,
                                                         half_t* __restrict__ Vt,
                                                         int B, int H, int S) {
  const size_t idx = (size_t)blockIdx.x * blockDim.x + threadIdx.x;
  const size_t total = (size_t)B * H * S * 64;
  if (idx >= total) return;
  const int d = (int)(idx & 63);
  size_t t = idx >> 6;
  const int s = (int)(t % S); t /= S;
  const int h = (int)(t % H);
  const int b = (int)(t / H);

  const size_t rowb = ((size_t)b * S + s) * 3072 + (size_t)h * 64;
  const float q = (float)qkv[rowb + d];
  const float k = (float)qkv[rowb + 1024 + d];
  const float v = (float)qkv[rowb + 2048 + d];
  const int   pd = (d < 32) ? d + 32 : d - 32;
  const float qp = (float)qkv[rowb + pd];
  const float kp = (float)qkv[rowb + 1024 + pd];

  const int fi = d & 31;
  const float inv_freq = __expf(-(float)(2 * fi) * (9.210340371976184f / 64.f));
  const float ang = (float)s * inv_freq;
  float si, c;
  __sincosf(ang, &si, &c);
  const float sgn = (d < 32) ? -1.f : 1.f;  // rotate_half sign

  const size_t o = ((size_t)(b * H + h) * S + s) * 64 + d;
  Qo[o] = (half_t)(q * c + sgn * qp * si);
  Ko[o] = (half_t)(k * c + sgn * kp * si);
  Vt[((size_t)(b * H + h) * 64 + d) * S + s] = (half_t)v;  // transposed
}

// ---- Flash attention: one wave per 16-query tile ----------------------------
__global__ __launch_bounds__(32) void flash_attn_kernel(const half_t* __restrict__ Q,
                                                        const half_t* __restrict__ Kh,
                                                        const half_t* __restrict__ Vt,
                                                        half_t* __restrict__ Out,
                                                        int B, int H, int S) {
  const int lane  = threadIdx.x;
  const int qbase = blockIdx.x * 16;
  const int h = blockIdx.y;
  const int b = blockIdx.z;
  const float scale = 0.125f;  // hd=64 -> 1/sqrt(64)

  const half_t* Qp = Q  + (size_t)(b * H + h) * S * 64;
  const half_t* Kp = Kh + (size_t)(b * H + h) * S * 64;
  const half_t* Vp = Vt + (size_t)(b * H + h) * 64 * S;  // [hd][S]

  __shared__ _Float16 Pl[16][32];  // C-layout -> A-fragment transpose buffer

  const v16h aq0 = load_a_frag(Qp, 64, qbase, 0,  lane);
  const v16h aq1 = load_a_frag(Qp, 64, qbase, 32, lane);

  v8f O0 = {}, O1 = {}, O2 = {}, O3 = {};
  float mrow[8], lrow[8];
#pragma unroll
  for (int r = 0; r < 8; ++r) { mrow[r] = -1e30f; lrow[r] = 0.f; }

  const int col = lane & 15;
  const int rh  = (lane >> 4) * 8;
  const int kend = qbase + 16;  // causal

  for (int kb = 0; kb < kend; kb += 32) {
    // ---- scores: S = Q K^T over hd=64 ----
    v8f s0 = {}, s1 = {};
    {
      v16h bk = load_b_frag(Kp, 64, kb, 0,  lane); s0 = WMMA_F16(aq0, bk, s0);
      bk      = load_b_frag(Kp, 64, kb, 32, lane); s0 = WMMA_F16(aq1, bk, s0);
      bk      = load_b_frag(Kp, 64, kb + 16, 0,  lane); s1 = WMMA_F16(aq0, bk, s1);
      bk      = load_b_frag(Kp, 64, kb + 16, 32, lane); s1 = WMMA_F16(aq1, bk, s1);
    }
    // ---- causal mask + scale, block row-max (16-lane half reduce) ----
    float sv0[8], sv1[8], bm[8];
#pragma unroll
    for (int r = 0; r < 8; ++r) {
      const int qr = qbase + rh + r;
      sv0[r] = (kb + col      <= qr) ? s0[r] * scale : -1e30f;
      sv1[r] = (kb + 16 + col <= qr) ? s1[r] * scale : -1e30f;
      bm[r]  = fmaxf(sv0[r], sv1[r]);
    }
#pragma unroll
    for (int off = 1; off < 16; off <<= 1)
#pragma unroll
      for (int r = 0; r < 8; ++r)
        bm[r] = fmaxf(bm[r], __shfl_xor(bm[r], off, 32));

    // ---- online softmax update + stage P into LDS as f16 ----
    float alpha[8], rs[8];
#pragma unroll
    for (int r = 0; r < 8; ++r) {
      const float mn = fmaxf(mrow[r], bm[r]);
      alpha[r] = __expf(mrow[r] - mn);
      mrow[r] = mn;
      const float p0 = __expf(sv0[r] - mn);
      const float p1 = __expf(sv1[r] - mn);
      Pl[rh + r][col]      = (half_t)p0;
      Pl[rh + r][col + 16] = (half_t)p1;
      rs[r] = p0 + p1;
    }
#pragma unroll
    for (int off = 1; off < 16; off <<= 1)
#pragma unroll
      for (int r = 0; r < 8; ++r)
        rs[r] += __shfl_xor(rs[r], off, 32);
#pragma unroll
    for (int r = 0; r < 8; ++r) {
      lrow[r] = lrow[r] * alpha[r] + rs[r];
      O0[r] *= alpha[r]; O1[r] *= alpha[r]; O2[r] *= alpha[r]; O3[r] *= alpha[r];
    }
    __syncthreads();

    // ---- reload P as A-fragment (16 rows x 32 keys) ----
    v16h pf;
    {
      const int m  = lane & 15;
      const int kk = (lane >> 4) * 16;
#pragma unroll
      for (int j = 0; j < 8; ++j) { pf[j] = Pl[m][kk + j]; pf[8 + j] = Pl[m][kk + 8 + j]; }
    }
    // ---- O += P V : V^T rows are K-contiguous -> clean B fragments ----
    {
      const int kk = (lane >> 4) * 16;
      const half_t* vp = Vp + kb + kk;
      v16h bv;
      bv = *(const v16h*)(vp + (size_t)(col)      * S); O0 = WMMA_F16(pf, bv, O0);
      bv = *(const v16h*)(vp + (size_t)(16 + col) * S); O1 = WMMA_F16(pf, bv, O1);
      bv = *(const v16h*)(vp + (size_t)(32 + col) * S); O2 = WMMA_F16(pf, bv, O2);
      bv = *(const v16h*)(vp + (size_t)(48 + col) * S); O3 = WMMA_F16(pf, bv, O3);
    }
    __syncthreads();
  }

  // ---- epilogue: normalize and write [B,S,D] f16 ----
  half_t* op = Out + ((size_t)b * S + qbase + rh) * 1024 + (size_t)h * 64;
#pragma unroll
  for (int r = 0; r < 8; ++r) {
    const float inv = 1.f / lrow[r];
    half_t* row = op + (size_t)r * 1024;
    row[col]      = (half_t)(O0[r] * inv);
    row[16 + col] = (half_t)(O1[r] * inv);
    row[32 + col] = (half_t)(O2[r] * inv);
    row[48 + col] = (half_t)(O3[r] * inv);
  }
}

// ---- host ------------------------------------------------------------------
extern "C" void kernel_launch(void* const* d_in, const int* in_sizes, int n_in,
                              void* d_out, int out_size, void* d_ws, size_t ws_size,
                              hipStream_t stream) {
  (void)in_sizes; (void)n_in; (void)out_size; (void)ws_size;
  const int B = 2, S = 2048, D = 1024, H = 16;

  const float* x     = (const float*)d_in[0];
  // d_in[1] = mask (implicit causal; unused)
  const float* w_qkv = (const float*)d_in[2];
  const float* w_out = (const float*)d_in[3];

  char* ws = (char*)d_ws;
  size_t off = 0;
  auto carve = [&](size_t bytes) -> void* {
    void* p = ws + off;
    off = (off + bytes + 255) & ~(size_t)255;
    return p;
  };
  half_t* x_h    = (half_t*)carve((size_t)B * S * D * 2);
  half_t* wqkv_h = (half_t*)carve((size_t)3 * D * D * 2);
  half_t* wout_h = (half_t*)carve((size_t)D * D * 2);
  half_t* qkv_h  = (half_t*)carve((size_t)B * S * 3 * D * 2);
  half_t* q_h    = (half_t*)carve((size_t)B * S * D * 2);
  half_t* k_h    = (half_t*)carve((size_t)B * S * D * 2);
  half_t* v_t    = (half_t*)carve((size_t)B * S * D * 2);
  half_t* att_h  = x_h;  // x_h is dead after the QKV GEMM; reuse its region

  // 1) f32 -> f16 operand conversion
  auto cvt = [&](const float* src, half_t* dst, size_t n) {
    f32_to_f16_kernel<<<(unsigned)((n + 255) / 256), 256, 0, stream>>>(src, dst, n);
  };
  cvt(x,     x_h,    (size_t)B * S * D);
  cvt(w_qkv, wqkv_h, (size_t)3 * D * D);
  cvt(w_out, wout_h, (size_t)D * D);

  // 2) QKV projection: [4096,1024] x [3072,1024]^T -> [4096,3072] f16
  {
    dim3 g((3 * D) / 128, (B * S) / 64);
    dim3 blk(32, 8);
    gemm_wmma_kernel<false><<<g, blk, 0, stream>>>(x_h, wqkv_h, qkv_h, B * S, 3 * D, D);
  }

  // 3) RoPE + split heads (+ V transpose)
  {
    size_t tot = (size_t)B * H * S * 64;
    rope_split_kernel<<<(unsigned)((tot + 255) / 256), 256, 0, stream>>>(qkv_h, q_h, k_h, v_t, B, H, S);
  }

  // 4) Causal flash attention (one wave per 16-query tile)
  {
    dim3 g(S / 16, H, B);
    flash_attn_kernel<<<g, 32, 0, stream>>>(q_h, k_h, v_t, att_h, B, H, S);
  }

  // 5) Output projection: [4096,1024] x [1024,1024]^T -> f32 d_out
  {
    dim3 g(D / 128, (B * S) / 64);
    dim3 blk(32, 8);
    gemm_wmma_kernel<true><<<g, blk, 0, stream>>>(att_h, wout_h, d_out, B * S, D, D);
  }
}
